// TokenDiffusion_43963285242224
// MI455X (gfx1250) — matrix operand order repaired and checked by reading, not compiled, not run on hardware
//
#include <hip/hip_runtime.h>
#include <hip/hip_bf16.h>
#include <stdint.h>

typedef __bf16 bf16;
typedef __attribute__((ext_vector_type(16))) __bf16 v16bf;
typedef __attribute__((ext_vector_type(8)))  __bf16 v8bf;
typedef __attribute__((ext_vector_type(8)))  float  v8f;

#define B_  2
#define N_  4096
#define D_  512
#define H_  8
#define DH_ 64
#define BN_ (B_ * N_)     // 8192 rows
#define BH_ (B_ * H_)     // 16 (b,h) pairs

// ---------------------------------------------------------------------------
// WMMA helpers (CDNA5 V_WMMA_F32_16X16X32_BF16, layouts per ISA 7.12.2)
// ---------------------------------------------------------------------------
__device__ __forceinline__ v8f wmma_bf16(v16bf a, v16bf b, v8f c) {
  // (neg_a, A, neg_b, B, c_mod, C, reuse_a, reuse_b)
  return __builtin_amdgcn_wmma_f32_16x16x32_bf16(false, a, false, b,
                                                 (short)0, c, false, false);
}

// A-matrix 16x32 (MxK), 16-bit: lanes 0-15 -> M=lane, K {0..7,16..23};
// lanes 16-31 -> M=lane-16, K {8..15,24..31}. Source row-major, ld elements.
__device__ __forceinline__ v16bf load_frag_a(const bf16* __restrict__ p,
                                             int ld, int row, int k, int lane) {
  int m  = lane & 15;
  int ko = (lane >> 4) * 8;
  const bf16* q = p + (size_t)(row + m) * ld + k + ko;
  union { v16bf v; v8bf h[2]; } u;
  u.h[0] = *(const v8bf*)(q);
  u.h[1] = *(const v8bf*)(q + 16);
  return u.v;
}

// Same pattern but from LDS (generic/ds path).
__device__ __forceinline__ v16bf load_frag_a_lds(const bf16* p, int ld, int lane) {
  int m  = lane & 15;
  int ko = (lane >> 4) * 8;
  const bf16* q = p + m * ld + ko;
  union { v16bf v; v8bf h[2]; } u;
  u.h[0] = *(const v8bf*)(q);
  u.h[1] = *(const v8bf*)(q + 16);
  return u.v;
}

// B-matrix 32x16 (KxN), 16-bit: lanes 0-15 -> N=lane, K {0..7,16..23};
// lanes 16-31 -> N=lane-16, K {8..15,24..31}.  B[k][n] = p[(col+n)*ld + kbase + k]
// (p holds the *transposed* operand row-major).
__device__ __forceinline__ v16bf load_frag_b(const bf16* __restrict__ p,
                                             int ld, int col, int k, int lane) {
  int n  = lane & 15;
  int ko = (lane >> 4) * 8;
  const bf16* q = p + (size_t)(col + n) * ld + k + ko;
  union { v16bf v; v8bf h[2]; } u;
  u.h[0] = *(const v8bf*)(q);
  u.h[1] = *(const v8bf*)(q + 16);
  return u.v;
}

// ---------------------------------------------------------------------------
// fp32 -> bf16 convert
// ---------------------------------------------------------------------------
__global__ void cvt_bf16_kernel(const float* __restrict__ in,
                                bf16* __restrict__ out, int n) {
  int i = blockIdx.x * blockDim.x + threadIdx.x;
  if (i < n) out[i] = (bf16)in[i];
}

// ---------------------------------------------------------------------------
// GEMM: C[M=8192, Nc=512] = X[8192,512] * W[512,512]^T   (bf16 in, f32 acc)
// mode 0: write qk head-major  (BH, N, DH)        bf16
// mode 1: write v  transposed  (BH, DH, N)        bf16
// mode 2: write plain fp32 row-major (final out)
// Block = (32,4): 4 waves, each wave a 64x64 tile; block covers 256 rows.
// Grid = (BN/256, 512/64) = (32, 8).
// ---------------------------------------------------------------------------
__global__ __launch_bounds__(128)
void proj_gemm_kernel(const bf16* __restrict__ X, const bf16* __restrict__ W,
                      bf16* __restrict__ outb, float* __restrict__ outf,
                      int mode) {
  const int lane  = threadIdx.x;
  const int wave  = threadIdx.y;
  const int mBase = blockIdx.x * 256 + wave * 64;
  const int nBase = blockIdx.y * 64;
  const int lrow  = lane & 15;
  const int hi    = lane >> 4;

  v8f acc[4][4] = {};

  for (int k = 0; k < D_; k += 32) {
    v16bf a[4], b[4];
#pragma unroll
    for (int mt = 0; mt < 4; ++mt)
      a[mt] = load_frag_a(X, D_, mBase + mt * 16, k, lane);
#pragma unroll
    for (int nt = 0; nt < 4; ++nt)
      b[nt] = load_frag_b(W, D_, nBase + nt * 16, k, lane);
#pragma unroll
    for (int mt = 0; mt < 4; ++mt)
#pragma unroll
      for (int nt = 0; nt < 4; ++nt)
        acc[mt][nt] = wmma_bf16(a[mt], b[nt], acc[mt][nt]);
  }

#pragma unroll
  for (int mt = 0; mt < 4; ++mt)
#pragma unroll
    for (int nt = 0; nt < 4; ++nt)
#pragma unroll
      for (int r = 0; r < 8; ++r) {
        float val = acc[mt][nt][r];
        int gm = mBase + mt * 16 + r + 8 * hi;   // global row in [0, BN)
        int gn = nBase + nt * 16 + lrow;         // global col in [0, D)
        if (mode == 2) {
          outf[(size_t)gm * D_ + gn] = val;
        } else {
          int bb = gm >> 12, n = gm & (N_ - 1);
          int h  = gn >> 6,  dh = gn & (DH_ - 1);
          if (mode == 0)       // qk head-major (BH, N, DH)
            outb[(((size_t)(bb * H_ + h)) * N_ + n) * DH_ + dh] = (bf16)val;
          else                 // v transposed  (BH, DH, N)
            outb[(((size_t)(bb * H_ + h)) * DH_ + dh) * N_ + n] = (bf16)val;
        }
      }
}

// ---------------------------------------------------------------------------
// q2[bh, n] = sum_dh qk[bh, n, dh]^2
// ---------------------------------------------------------------------------
__global__ void q2_kernel(const bf16* __restrict__ qkh, float* __restrict__ q2) {
  int row = blockIdx.x * blockDim.x + threadIdx.x;   // [0, BH*N)
  const bf16* p = qkh + (size_t)row * DH_;
  float s = 0.f;
#pragma unroll
  for (int c = 0; c < DH_ / 8; ++c) {
    v8bf v = *(const v8bf*)(p + c * 8);
#pragma unroll
    for (int i = 0; i < 8; ++i) { float f = (float)v[i]; s += f * f; }
  }
  q2[row] = s;
}

// ---------------------------------------------------------------------------
// Flash attention (tied Q=K, distance logits, per-head tau).
// Block = (32,4): each wave owns 16 queries; block covers 64.
// Grid = (N/64, BH). Keys streamed in chunks of 32 with online softmax.
// ---------------------------------------------------------------------------
__global__ __launch_bounds__(128)
void attn_kernel(const bf16* __restrict__ qkh,   // (BH, N, DH)
                 const bf16* __restrict__ vt,    // (BH, DH, N)
                 const float* __restrict__ q2,   // (BH, N)
                 const float* __restrict__ tau,  // (H)
                 bf16* __restrict__ att) {       // (B, N, D) bf16
  __shared__ __align__(16) bf16 lds[4][16 * 48]; // per-wave 16x32 P tile, stride 48

  const int lane  = threadIdx.x;
  const int wave  = threadIdx.y;
  const int bh    = blockIdx.y;
  const int h     = bh & (H_ - 1);
  const int bb    = bh >> 3;
  const int qBase = blockIdx.x * 64 + wave * 16;
  const int lrow  = lane & 15;
  const int hi    = lane >> 4;

  const bf16*  qk  = qkh + (size_t)bh * N_ * DH_;
  const bf16*  v   = vt  + (size_t)bh * DH_ * N_;
  const float* q2h = q2  + (size_t)bh * N_;

  // fold tau/sqrt(dh) and log2(e) into one scale; work in base-2 exponentials
  const float scale  = tau[h] * (1.4426950408889634f / 8.0f);
  const float dscale = 2.0f * scale;

  // Q operand, K-dim = 64 -> two A fragments, loaded once
  v16bf aq0 = load_frag_a(qk, DH_, qBase, 0,  lane);
  v16bf aq1 = load_frag_a(qk, DH_, qBase, 32, lane);

  float mrow[8], lsum[8], q2i[8];
#pragma unroll
  for (int r = 0; r < 8; ++r) {
    mrow[r] = -3.0e38f;
    lsum[r] = 0.f;
    q2i[r]  = q2h[qBase + r + 8 * hi] * scale;
  }
  v8f o[4] = {};

  bf16* myP = lds[wave];

  for (int j0 = 0; j0 < N_; j0 += 32) {
    // ---- S = Q * K^T for 16x32 key slab (two 16x16 tiles) ----
    v8f s[2] = {};
#pragma unroll
    for (int jt = 0; jt < 2; ++jt) {
      v16bf bk0 = load_frag_b(qk, DH_, j0 + jt * 16, 0,  lane);
      v16bf bk1 = load_frag_b(qk, DH_, j0 + jt * 16, 32, lane);
      s[jt] = wmma_bf16(aq0, bk0, s[jt]);
      s[jt] = wmma_bf16(aq1, bk1, s[jt]);
    }
    float q2j0 = q2h[j0 + lrow] * scale;
    float q2j1 = q2h[j0 + 16 + lrow] * scale;

    // ---- logits + online softmax stats ----
    float nm[8], alpha[8];
#pragma unroll
    for (int r = 0; r < 8; ++r) {
      float e0 = dscale * s[0][r] - q2i[r] - q2j0;
      float e1 = dscale * s[1][r] - q2i[r] - q2j1;
      s[0][r] = e0; s[1][r] = e1;
      float mx = fmaxf(e0, e1);
#pragma unroll
      for (int msk = 8; msk >= 1; msk >>= 1)
        mx = fmaxf(mx, __shfl_xor(mx, msk, 16));
      nm[r] = fmaxf(mrow[r], mx);
    }
#pragma unroll
    for (int r = 0; r < 8; ++r) {
      float p0 = exp2f(s[0][r] - nm[r]);
      float p1 = exp2f(s[1][r] - nm[r]);
      s[0][r] = p0; s[1][r] = p1;
      float rs = p0 + p1;
#pragma unroll
      for (int msk = 8; msk >= 1; msk >>= 1)
        rs += __shfl_xor(rs, msk, 16);
      alpha[r] = exp2f(mrow[r] - nm[r]);
      lsum[r]  = lsum[r] * alpha[r] + rs;
      mrow[r]  = nm[r];
    }
#pragma unroll
    for (int t = 0; t < 4; ++t)
#pragma unroll
      for (int r = 0; r < 8; ++r) o[t][r] *= alpha[r];

    // ---- repack P (D-layout) -> A-layout via per-wave LDS tile ----
#pragma unroll
    for (int r = 0; r < 8; ++r) {
      int row = r + 8 * hi;
      myP[row * 48 + lrow]      = (bf16)s[0][r];
      myP[row * 48 + 16 + lrow] = (bf16)s[1][r];
    }
    v16bf pa = load_frag_a_lds(myP, 48, lane);

    // ---- O += P * V (K = 32 keys, four dh tiles) ----
#pragma unroll
    for (int t = 0; t < 4; ++t) {
      v16bf vb = load_frag_b(v, N_, t * 16, j0, lane);
      o[t] = wmma_bf16(pa, vb, o[t]);
    }
  }

  // ---- finalize: divide by row sums, write merged-head bf16 activation ----
#pragma unroll
  for (int r = 0; r < 8; ++r) lsum[r] = 1.0f / lsum[r];
#pragma unroll
  for (int t = 0; t < 4; ++t)
#pragma unroll
    for (int r = 0; r < 8; ++r) {
      int n  = qBase + r + 8 * hi;
      int dh = t * 16 + lrow;
      att[((size_t)(bb * N_ + n)) * D_ + h * DH_ + dh] = (bf16)(o[t][r] * lsum[r]);
    }
}

// ---------------------------------------------------------------------------
// Launch
// ---------------------------------------------------------------------------
extern "C" void kernel_launch(void* const* d_in, const int* in_sizes, int n_in,
                              void* d_out, int out_size, void* d_ws, size_t ws_size,
                              hipStream_t stream) {
  const float* x   = (const float*)d_in[0];
  const float* Wqk = (const float*)d_in[1];
  const float* Wv  = (const float*)d_in[2];
  const float* Wo  = (const float*)d_in[3];
  const float* tau = (const float*)d_in[4];
  float* out = (float*)d_out;

  // workspace carve-up (bytes)
  char* w = (char*)d_ws;
  const size_t SZ_X  = (size_t)BN_ * D_ * sizeof(bf16);      // 8 MiB
  const size_t SZ_W  = (size_t)D_ * D_ * sizeof(bf16);       // 512 KiB
  const size_t SZ_Q2 = (size_t)BH_ * N_ * sizeof(float);     // 256 KiB
  bf16*  xb   = (bf16*)(w);
  bf16*  wqkb = (bf16*)(w + SZ_X);
  bf16*  wvb  = (bf16*)(w + SZ_X + SZ_W);
  bf16*  wob  = (bf16*)(w + SZ_X + 2 * SZ_W);
  bf16*  qkh  = (bf16*)(w + SZ_X + 3 * SZ_W);
  bf16*  vt   = (bf16*)(w + 2 * SZ_X + 3 * SZ_W);
  float* q2   = (float*)(w + 3 * SZ_X + 3 * SZ_W);
  bf16*  att  = (bf16*)(w + 3 * SZ_X + 3 * SZ_W + SZ_Q2);

  // 1) convert fp32 -> bf16
  int nx = BN_ * D_, nw = D_ * D_;
  cvt_bf16_kernel<<<(nx + 255) / 256, 256, 0, stream>>>(x,   xb,   nx);
  cvt_bf16_kernel<<<(nw + 255) / 256, 256, 0, stream>>>(Wqk, wqkb, nw);
  cvt_bf16_kernel<<<(nw + 255) / 256, 256, 0, stream>>>(Wv,  wvb,  nw);
  cvt_bf16_kernel<<<(nw + 255) / 256, 256, 0, stream>>>(Wo,  wob,  nw);

  // 2) projections (WMMA GEMMs)
  dim3 blk(32, 4);
  dim3 gproj(BN_ / 256, D_ / 64);
  proj_gemm_kernel<<<gproj, blk, 0, stream>>>(xb, wqkb, qkh, nullptr, 0);
  proj_gemm_kernel<<<gproj, blk, 0, stream>>>(xb, wvb,  vt,  nullptr, 1);

  // 3) |q|^2 row sums
  q2_kernel<<<(BH_ * N_) / 256, 256, 0, stream>>>(qkh, q2);

  // 4) flash attention
  attn_kernel<<<dim3(N_ / 64, BH_), blk, 0, stream>>>(qkh, vt, q2, tau, att);

  // 5) output projection -> fp32
  proj_gemm_kernel<<<gproj, blk, 0, stream>>>(att, wob, nullptr, out, 2);
}
